// BasicTransformerBlock_6253472383697
// MI455X (gfx1250) — matrix operand order repaired and checked by reading, not compiled, not run on hardware
//
#include <hip/hip_runtime.h>
#include <hip/hip_bf16.h>
#include <math.h>

typedef _Float16 half_t;
typedef __attribute__((ext_vector_type(16))) _Float16 v16h;
typedef __attribute__((ext_vector_type(8)))  _Float16 v8h;
typedef __attribute__((ext_vector_type(8)))  float    v8f;
typedef __attribute__((ext_vector_type(4)))  int      v4i;

#define BF    16      // batch*frames
#define DSP   576     // spatial tokens per frame
#define DIMC  320     // channel dim
#define NHEAD 8
#define DH    40
#define DHP   64      // padded head dim
#define CROSSD 768
#define TLEN  77
#define ROWS  (BF*DSP)   // 9216
#define FFI   1280
#define ATTN_SCALE 0.15811388300841897f   // 40^-0.5

union F16frag { v16h v; v8h h[2]; };

// ---- gfx1250 async global->LDS staging (guarded; fallback = load + ds_store) ----
#if defined(__has_builtin)
#if __has_builtin(__builtin_amdgcn_global_load_async_to_lds_b128) && \
    __has_builtin(__builtin_amdgcn_s_wait_asynccnt)
#define USE_ASYNC_LDS 1
#endif
#endif
#ifndef USE_ASYNC_LDS
#define USE_ASYNC_LDS 0
#endif

#define AS1 __attribute__((address_space(1)))
#define AS3 __attribute__((address_space(3)))

__device__ __forceinline__ void stage16(const half_t* g, half_t* l) {
#if USE_ASYNC_LDS
  __builtin_amdgcn_global_load_async_to_lds_b128((AS1 v4i*)g, (AS3 v4i*)l, 0, 0);
#else
  *(v8h*)l = *(const v8h*)g;
#endif
}
template <int N>
__device__ __forceinline__ void async_wait() {
#if USE_ASYNC_LDS
  __builtin_amdgcn_s_wait_asynccnt(N);
#endif
}

// ---------------- weight transpose + f16 convert:  Wt[n*K+k] = W[k*N+n] ----------------
__global__ void k_wt(const float* __restrict__ W, half_t* __restrict__ Wt, int K, int N) {
  int i = blockIdx.x * 256 + threadIdx.x;
  if (i >= K * N) return;
  int n = i / K, k = i - n * K;
  Wt[i] = (half_t)W[(size_t)k * N + n];
}

// ---------------- f32 -> f16 copy ----------------
__global__ void k_cvt(const float* __restrict__ x, half_t* __restrict__ y, int n) {
  int i = blockIdx.x * 256 + threadIdx.x;
  if (i < n) y[i] = (half_t)x[i];
}

// ---------------- LayerNorm over last dim (320), one wave32 per row, f16 out ----------------
__global__ void k_ln(const float* __restrict__ x, const float* __restrict__ g,
                     const float* __restrict__ b, half_t* __restrict__ y, int rows) {
  int row  = blockIdx.x * 8 + (threadIdx.x >> 5);
  int lane = threadIdx.x & 31;
  if (row >= rows) return;
  const float* xr = x + (size_t)row * DIMC;
  float s = 0.f, s2 = 0.f;
  for (int i = lane; i < DIMC; i += 32) { float v = xr[i]; s += v; s2 += v * v; }
#pragma unroll
  for (int o = 16; o; o >>= 1) { s += __shfl_xor(s, o, 32); s2 += __shfl_xor(s2, o, 32); }
  float mu  = s * (1.f / DIMC);
  float var = fmaxf(s2 * (1.f / DIMC) - mu * mu, 0.f);
  float inv = rsqrtf(var + 1e-5f);
  half_t* yr = y + (size_t)row * DIMC;
  for (int i = lane; i < DIMC; i += 32) yr[i] = (half_t)((xr[i] - mu) * inv * g[i] + b[i]);
}

// ---------------- generic WMMA GEMM: C[M,N] = A[M,K](f16) @ Bt[N,K](f16)^T ----------------
// 4 waves / WG; wave w -> rows [m0+16w,+16), cols [n0,n0+64). B tile double-buffered in LDS,
// staged with async global->LDS (tile i+1 in flight while WMMAs consume tile i).
template <bool OUTF16>
__global__ void k_gemm(const half_t* __restrict__ A, const half_t* __restrict__ Bt,
                       half_t* __restrict__ outh, float* __restrict__ outf,
                       const float* __restrict__ bias, const float* __restrict__ resid,
                       int M, int N, int K) {
  __shared__ __align__(16) half_t ldsB[2][64 * 40];
  const int lane = threadIdx.x & 31;
  const int wave = threadIdx.x >> 5;
  const int lm = lane & 15;
  const int hi = lane >> 4;
  const int kbA = hi * 16;   // A-frag K base: contiguous 16
  const int kbB = hi * 16;   // B-frag K base: contiguous 16
  const int m0 = blockIdx.x * 64 + wave * 16;
  const int n0 = blockIdx.y * 64;
  v8f acc[4] = {};

  int mrow = m0 + lm; if (mrow >= M) mrow = M - 1;   // clamp loads; stores guarded
  const half_t* Ar = A + (size_t)mrow * K;

  const int srow = threadIdx.x >> 1;            // 0..63
  const int sseg = (threadIdx.x & 1) * 16;      // 0 or 16
  const half_t* Bg = Bt + (size_t)(n0 + srow) * K + sseg;
  half_t* Ld0 = &ldsB[0][srow * 40 + sseg];
  half_t* Ld1 = &ldsB[1][srow * 40 + sseg];

  stage16(Bg, Ld0);                              // prologue: tile 0 -> buf 0
  stage16(Bg + 8, Ld0 + 8);

  const int nT = K >> 5;
  for (int i = 0; i < nT; ++i) {
    const int k0 = i << 5;
    half_t* buf = (i & 1) ? &ldsB[1][0] : &ldsB[0][0];
    if (i + 1 < nT) {                            // stage next tile into other buffer
      half_t* nb = (i & 1) ? Ld0 : Ld1;
      stage16(Bg + k0 + 32, nb);
      stage16(Bg + k0 + 40, nb + 8);
      async_wait<2>();                           // current tile's 2 async ops complete
    } else {
      async_wait<0>();
    }
    __syncthreads();                             // tile i visible to all waves
    if (k0 + 64 < K) __builtin_prefetch((const void*)(Ar + k0 + 64), 0, 1);
    F16frag af;
    af.h[0] = *(const v8h*)(Ar + k0 + kbA);
    af.h[1] = *(const v8h*)(Ar + k0 + kbA + 8);
#pragma unroll
    for (int j = 0; j < 4; ++j) {
      F16frag bf;
      const half_t* bp = &buf[(j * 16 + lm) * 40 + kbB];
      bf.h[0] = *(const v8h*)bp;
      bf.h[1] = *(const v8h*)(bp + 8);
      acc[j] = __builtin_amdgcn_wmma_f32_16x16x32_f16(false, af.v, false, bf.v,
                                                      (short)0, acc[j], false, false);
    }
    __syncthreads();                             // done reading buf before it is re-staged
  }
  const int mh = hi * 8;
#pragma unroll
  for (int j = 0; j < 4; ++j) {
    int n = n0 + j * 16 + lm;
#pragma unroll
    for (int r = 0; r < 8; ++r) {
      int m = m0 + mh + r;
      if (m >= M) continue;
      float v = acc[j][r];
      if (bias)  v += bias[n];
      if (resid) v += resid[(size_t)m * N + n];
      if (OUTF16) outh[(size_t)m * N + n] = (half_t)v;
      else        outf[(size_t)m * N + n] = v;
    }
  }
}

// ---------------- GEGLU GEMM: proj = A @ Wt + b ; out = x1 * gelu(gate) (f16) ----------------
__global__ void k_geglu(const half_t* __restrict__ A, const half_t* __restrict__ Bt,
                        const float* __restrict__ bias, half_t* __restrict__ out) {
  __shared__ __align__(16) half_t ldsB[2][128 * 40];
  const int lane = threadIdx.x & 31;
  const int wave = threadIdx.x >> 5;
  const int lm = lane & 15, hi = lane >> 4;
  const int kbA = hi * 16, kbB = hi * 16;
  const int m0 = blockIdx.x * 64 + wave * 16;
  const int n0 = blockIdx.y * 64;
  v8f ax[4] = {}, ag[4] = {};
  const half_t* Ar = A + (size_t)(m0 + lm) * DIMC;

  const int srow = threadIdx.x;  // 0..127: first 64 rows = x1 cols, next 64 = gate cols
  const int grow = (srow < 64) ? (n0 + srow) : (FFI + n0 + srow - 64);
  const half_t* Bg = Bt + (size_t)grow * DIMC;
  half_t* Ld0 = &ldsB[0][srow * 40];
  half_t* Ld1 = &ldsB[1][srow * 40];

#pragma unroll
  for (int q = 0; q < 4; ++q) stage16(Bg + q * 8, Ld0 + q * 8);   // prologue tile 0

  const int nT = DIMC >> 5;   // 10
  for (int i = 0; i < nT; ++i) {
    const int k0 = i << 5;
    half_t* buf = (i & 1) ? &ldsB[1][0] : &ldsB[0][0];
    if (i + 1 < nT) {
      half_t* nb = (i & 1) ? Ld0 : Ld1;
#pragma unroll
      for (int q = 0; q < 4; ++q) stage16(Bg + k0 + 32 + q * 8, nb + q * 8);
      async_wait<4>();
    } else {
      async_wait<0>();
    }
    __syncthreads();
    if (k0 + 64 < DIMC) __builtin_prefetch((const void*)(Ar + k0 + 64), 0, 1);
    F16frag af;
    af.h[0] = *(const v8h*)(Ar + k0 + kbA);
    af.h[1] = *(const v8h*)(Ar + k0 + kbA + 8);
#pragma unroll
    for (int j = 0; j < 4; ++j) {
      F16frag bx, bg;
      const half_t* px = &buf[(j * 16 + lm) * 40 + kbB];
      const half_t* pg = &buf[(64 + j * 16 + lm) * 40 + kbB];
      bx.h[0] = *(const v8h*)px; bx.h[1] = *(const v8h*)(px + 8);
      bg.h[0] = *(const v8h*)pg; bg.h[1] = *(const v8h*)(pg + 8);
      ax[j] = __builtin_amdgcn_wmma_f32_16x16x32_f16(false, af.v, false, bx.v,
                                                     (short)0, ax[j], false, false);
      ag[j] = __builtin_amdgcn_wmma_f32_16x16x32_f16(false, af.v, false, bg.v,
                                                     (short)0, ag[j], false, false);
    }
    __syncthreads();
  }
  const int mh = hi * 8;
#pragma unroll
  for (int j = 0; j < 4; ++j) {
    int n = n0 + j * 16 + lm;
#pragma unroll
    for (int r = 0; r < 8; ++r) {
      int m = m0 + mh + r;
      float xv = ax[j][r] + bias[n];
      float gv = ag[j][r] + bias[FFI + n];
      float ge = 0.5f * gv * (1.f + erff(gv * 0.70710678118654752f));
      out[(size_t)m * FFI + n] = (half_t)(xv * ge);
    }
  }
}

// ---------------- repack: compact [rows,320] f16 -> per-head padded layouts ----------------
// mode 0: self/cross Q    (row = u*576 + s)
// mode 1: sparse-causal KV (s<576 -> frame0 row s ; else frame max(u-1,0) row s-576)
// mode 2: cross KV        (row = u*77 + s, s>=77 zero-padded)
// mode 3: temporal        (row = s*576 + u)
// isV=0: dst[((u*8+h)*spad+s)*64+dc]   isV=1: dst[((u*8+h)*64+dc)*spad+s]
__global__ void k_repack(const half_t* __restrict__ src, half_t* __restrict__ dst,
                         int mode, int isV, int units, int spad, int slen) {
  size_t idx = (size_t)blockIdx.x * 256 + threadIdx.x;
  size_t total = (size_t)units * NHEAD * spad * DHP;
  if (idx >= total) return;
  int dc = (int)(idx & 63);
  size_t t = idx >> 6;
  int s = (int)(t % spad); t /= spad;
  int h = (int)(t % NHEAD);
  int u = (int)(t / NHEAD);
  half_t val = (half_t)0.f;
  if (dc < DH && s < slen) {
    int row;
    if (mode == 0)      row = u * DSP + s;
    else if (mode == 1) { int f = (s < DSP) ? 0 : ((u > 0) ? u - 1 : 0);
                          int ss = (s < DSP) ? s : s - DSP; row = f * DSP + ss; }
    else if (mode == 2) row = u * TLEN + s;
    else                row = s * DSP + u;
    val = src[(size_t)row * DIMC + h * DH + dc];
  }
  if (isV) dst[(((size_t)u * NHEAD + h) * DHP + dc) * spad + s] = val;
  else     dst[(((size_t)u * NHEAD + h) * spad + s) * DHP + dc] = val;
}

// ---------------- flash attention: one wave32 per 16-query tile ----------------
// Qp: [uh][sq][64], Kp: [uh][spad][64], Vt: [uh][64][spad]; out compact f16 [rows,320]
__global__ void k_attn(const half_t* __restrict__ Qp, const half_t* __restrict__ Kp,
                       const half_t* __restrict__ Vt, half_t* __restrict__ outO,
                       int sq, int spad, int slen, float scale, int temporal) {
  __shared__ __align__(16) half_t Pld[16 * 72];
  const int uh = blockIdx.x;
  const int qt = blockIdx.y;
  const int lane = threadIdx.x;
  const int lm = lane & 15;
  const int hi = lane >> 4;
  const int kbA = hi * 8;
  const int kbB = hi * 8;

  for (int i = lane; i < 16 * 72; i += 32) Pld[i] = (half_t)0.f;

  const half_t* Qb = Qp + ((size_t)uh * sq + qt * 16) * DHP;
  const half_t* Kb = Kp + (size_t)uh * spad * DHP;
  const half_t* Vb = Vt + (size_t)uh * DHP * spad;

  F16frag qf[2];
#pragma unroll
  for (int t = 0; t < 2; ++t) {
    const half_t* p = Qb + (size_t)lm * DHP + t * 32 + kbA;
    qf[t].h[0] = *(const v8h*)p;
    qf[t].h[1] = *(const v8h*)(p + 16);
  }

  v8f o[4] = {};
  float mrow = -1e30f, lrow = 0.f;

  for (int kb0 = 0; kb0 < spad; kb0 += 64) {
    if (kb0 + 64 < spad) {   // hint next key/value block toward the caches
      __builtin_prefetch((const void*)(Kb + (size_t)(kb0 + 64 + lm) * DHP), 0, 1);
      __builtin_prefetch((const void*)(Vb + (size_t)lm * spad + kb0 + 64), 0, 1);
    }
    v8f s[4] = {};
#pragma unroll
    for (int j = 0; j < 4; ++j) {
      int key = kb0 + j * 16 + lm;
#pragma unroll
      for (int t = 0; t < 2; ++t) {
        F16frag kf;
        const half_t* p = Kb + (size_t)key * DHP + t * 32 + kbB;
        kf.h[0] = *(const v8h*)p;
        kf.h[1] = *(const v8h*)(p + 16);
        s[j] = __builtin_amdgcn_wmma_f32_16x16x32_f16(false, qf[t].v, false, kf.v,
                                                      (short)0, s[j], false, false);
      }
    }
    // scale + key mask + tile row-max
    float tmax = -1e30f;
#pragma unroll
    for (int j = 0; j < 4; ++j) {
      bool valid = (kb0 + j * 16 + lm) < slen;
#pragma unroll
      for (int r = 0; r < 8; ++r) {
        float v = valid ? s[j][r] * scale : -1e30f;
        s[j][r] = v;
        tmax = fmaxf(tmax, v);
      }
    }
#pragma unroll
    for (int off = 1; off < 16; off <<= 1) tmax = fmaxf(tmax, __shfl_xor(tmax, off, 32));
    float mnew  = fmaxf(mrow, tmax);
    float alpha = __expf(mrow - mnew);
    lrow *= alpha;
#pragma unroll
    for (int j = 0; j < 4; ++j)
#pragma unroll
      for (int r = 0; r < 8; ++r) o[j][r] *= alpha;

    float ps = 0.f;
#pragma unroll
    for (int j = 0; j < 4; ++j) {
#pragma unroll
      for (int r = 0; r < 8; ++r) {
        float p = __expf(s[j][r] - mnew);
        ps += p;
        Pld[(r + hi * 8) * 72 + j * 16 + lm] = (half_t)p;   // C-layout -> row-major LDS
      }
    }
#pragma unroll
    for (int off = 1; off < 16; off <<= 1) ps += __shfl_xor(ps, off, 32);
    lrow += ps;
    mrow = mnew;

    // P @ V  (read P back in A-fragment layout)
#pragma unroll
    for (int t = 0; t < 2; ++t) {
      F16frag pf;
      const half_t* pp = &Pld[lm * 72 + t * 32 + kbA];
      pf.h[0] = *(const v8h*)pp;
      pf.h[1] = *(const v8h*)(pp + 16);
#pragma unroll
      for (int j = 0; j < 4; ++j) {
        F16frag vf;
        const half_t* vp = Vb + (size_t)(j * 16 + lm) * spad + kb0 + t * 32 + kbB;
        vf.h[0] = *(const v8h*)vp;
        vf.h[1] = *(const v8h*)(vp + 16);
        o[j] = __builtin_amdgcn_wmma_f32_16x16x32_f16(false, pf.v, false, vf.v,
                                                      (short)0, o[j], false, false);
      }
    }
  }
  float linv = 1.f / lrow;
  int u = uh >> 3, h = uh & 7;
#pragma unroll
  for (int j = 0; j < 4; ++j) {
    int dc = j * 16 + lm;
    if (dc >= DH) continue;
#pragma unroll
    for (int r = 0; r < 8; ++r) {
      int q = qt * 16 + r + hi * 8;
      size_t row = temporal ? ((size_t)q * DSP + u) : ((size_t)u * sq + q);
      outO[row * DIMC + h * DH + dc] = (half_t)(o[j][r] * linv);
    }
  }
}

// =====================================================================================
extern "C" void kernel_launch(void* const* d_in, const int* in_sizes, int n_in,
                              void* d_out, int out_size, void* d_ws, size_t ws_size,
                              hipStream_t stream) {
  (void)in_sizes; (void)n_in; (void)out_size; (void)ws_size;

  const float* h0    = (const float*)d_in[0];
  const float* enc   = (const float*)d_in[1];
  const float* ln1_w = (const float*)d_in[2];  const float* ln1_b = (const float*)d_in[3];
  const float* q1_w  = (const float*)d_in[4];  const float* k1_w  = (const float*)d_in[5];
  const float* v1_w  = (const float*)d_in[6];  const float* o1_w  = (const float*)d_in[7];
  const float* o1_b  = (const float*)d_in[8];
  const float* ln2_w = (const float*)d_in[9];  const float* ln2_b = (const float*)d_in[10];
  const float* q2_w  = (const float*)d_in[11]; const float* k2_w  = (const float*)d_in[12];
  const float* v2_w  = (const float*)d_in[13]; const float* o2_w  = (const float*)d_in[14];
  const float* o2_b  = (const float*)d_in[15];
  const float* lnt_w = (const float*)d_in[16]; const float* lnt_b = (const float*)d_in[17];
  const float* qt_w  = (const float*)d_in[18]; const float* kt_w  = (const float*)d_in[19];
  const float* vt_w  = (const float*)d_in[20]; const float* ot_w  = (const float*)d_in[21];
  const float* ot_b  = (const float*)d_in[22];
  const float* ln3_w = (const float*)d_in[23]; const float* ln3_b = (const float*)d_in[24];
  const float* gg_w  = (const float*)d_in[25]; const float* gg_b  = (const float*)d_in[26];
  const float* ffo_w = (const float*)d_in[27]; const float* ffo_b = (const float*)d_in[28];
  float* out = (float*)d_out;

  // ---- workspace carve-up (deterministic) ----
  char* ws = (char*)d_ws;
  size_t off = 0;
  auto carve = [&](size_t elems, size_t esz) {
    size_t o = off; off += ((elems * esz + 255) / 256) * 256; return (void*)(ws + o);
  };
  half_t* wq1t = (half_t*)carve(320 * 320, 2);
  half_t* wk1t = (half_t*)carve(320 * 320, 2);
  half_t* wv1t = (half_t*)carve(320 * 320, 2);
  half_t* wo1t = (half_t*)carve(320 * 320, 2);
  half_t* wq2t = (half_t*)carve(320 * 320, 2);
  half_t* wk2t = (half_t*)carve((size_t)320 * CROSSD, 2);
  half_t* wv2t = (half_t*)carve((size_t)320 * CROSSD, 2);
  half_t* wo2t = (half_t*)carve(320 * 320, 2);
  half_t* wqtt = (half_t*)carve(320 * 320, 2);
  half_t* wktt = (half_t*)carve(320 * 320, 2);
  half_t* wvtt = (half_t*)carve(320 * 320, 2);
  half_t* wott = (half_t*)carve(320 * 320, 2);
  half_t* wggt = (half_t*)carve((size_t)2 * FFI * DIMC, 2);
  half_t* wfft = (half_t*)carve((size_t)DIMC * FFI, 2);
  half_t* enc16  = (half_t*)carve((size_t)BF * TLEN * CROSSD, 2);
  half_t* norm16 = (half_t*)carve((size_t)ROWS * DIMC, 2);
  half_t* q16    = (half_t*)carve((size_t)ROWS * DIMC, 2);
  half_t* k16    = (half_t*)carve((size_t)ROWS * DIMC, 2);
  half_t* v16    = (half_t*)carve((size_t)ROWS * DIMC, 2);
  half_t* Qp     = (half_t*)carve((size_t)BF * NHEAD * DSP * DHP, 2);
  half_t* Kp     = (half_t*)carve((size_t)DSP * NHEAD * DHP * DHP, 2);   // max arena
  half_t* Vt     = (half_t*)carve((size_t)DSP * NHEAD * DHP * DHP, 2);
  half_t* aO16   = (half_t*)carve((size_t)ROWS * DIMC, 2);
  float*  hA     = (float*)carve((size_t)ROWS * DIMC, 4);
  float*  hB     = (float*)carve((size_t)ROWS * DIMC, 4);
  half_t* ffin   = Kp;   // stage-4 reuse of Kp arena (23.6MB <= 37.7MB)

  auto T = [&](const float* W, half_t* Wt, int K, int N) {
    int n = K * N;
    k_wt<<<(n + 255) / 256, 256, 0, stream>>>(W, Wt, K, N);
  };
  auto GEMMh = [&](const half_t* A, const half_t* Bt, half_t* O, int M, int N, int K) {
    dim3 g((M + 63) / 64, N / 64);
    k_gemm<true><<<g, 128, 0, stream>>>(A, Bt, O, nullptr, nullptr, nullptr, M, N, K);
  };
  auto GEMMf = [&](const half_t* A, const half_t* Bt, float* O, const float* bias,
                   const float* res, int M, int N, int K) {
    dim3 g((M + 63) / 64, N / 64);
    k_gemm<false><<<g, 128, 0, stream>>>(A, Bt, nullptr, O, bias, res, M, N, K);
  };
  auto REP = [&](const half_t* s, half_t* d, int mode, int isV, int units, int spad, int slen) {
    size_t tot = (size_t)units * NHEAD * spad * DHP;
    k_repack<<<(unsigned)((tot + 255) / 256), 256, 0, stream>>>(s, d, mode, isV, units, spad, slen);
  };
  auto LN = [&](const float* x, const float* g, const float* b, half_t* y) {
    k_ln<<<ROWS / 8, 256, 0, stream>>>(x, g, b, y, ROWS);
  };

  // ---- one-time per launch: weight transpose/convert ----
  T(q1_w, wq1t, DIMC, DIMC);  T(k1_w, wk1t, DIMC, DIMC);
  T(v1_w, wv1t, DIMC, DIMC);  T(o1_w, wo1t, DIMC, DIMC);
  T(q2_w, wq2t, DIMC, DIMC);  T(k2_w, wk2t, CROSSD, DIMC);
  T(v2_w, wv2t, CROSSD, DIMC); T(o2_w, wo2t, DIMC, DIMC);
  T(qt_w, wqtt, DIMC, DIMC);  T(kt_w, wktt, DIMC, DIMC);
  T(vt_w, wvtt, DIMC, DIMC);  T(ot_w, wott, DIMC, DIMC);
  T(gg_w, wggt, DIMC, 2 * FFI); T(ffo_w, wfft, FFI, DIMC);
  {
    int n = BF * TLEN * CROSSD;
    k_cvt<<<(n + 255) / 256, 256, 0, stream>>>(enc, enc16, n);
  }

  // ---- stage 1: sparse-causal self attention ----
  LN(h0, ln1_w, ln1_b, norm16);
  GEMMh(norm16, wq1t, q16, ROWS, DIMC, DIMC);
  GEMMh(norm16, wk1t, k16, ROWS, DIMC, DIMC);
  GEMMh(norm16, wv1t, v16, ROWS, DIMC, DIMC);
  REP(q16, Qp, 0, 0, BF, DSP, DSP);
  REP(k16, Kp, 1, 0, BF, 2 * DSP, 2 * DSP);
  REP(v16, Vt, 1, 1, BF, 2 * DSP, 2 * DSP);
  k_attn<<<dim3(BF * NHEAD, DSP / 16), 32, 0, stream>>>(Qp, Kp, Vt, aO16,
                                                        DSP, 2 * DSP, 2 * DSP, ATTN_SCALE, 0);
  GEMMf(aO16, wo1t, hA, o1_b, h0, ROWS, DIMC, DIMC);

  // ---- stage 2: cross attention ----
  LN(hA, ln2_w, ln2_b, norm16);
  GEMMh(norm16, wq2t, q16, ROWS, DIMC, DIMC);
  GEMMh(enc16, wk2t, k16, BF * TLEN, DIMC, CROSSD);
  GEMMh(enc16, wv2t, v16, BF * TLEN, DIMC, CROSSD);
  REP(q16, Qp, 0, 0, BF, DSP, DSP);
  REP(k16, Kp, 2, 0, BF, 128, TLEN);
  REP(v16, Vt, 2, 1, BF, 128, TLEN);
  k_attn<<<dim3(BF * NHEAD, DSP / 16), 32, 0, stream>>>(Qp, Kp, Vt, aO16,
                                                        DSP, 128, TLEN, ATTN_SCALE, 0);
  GEMMf(aO16, wo2t, hB, o2_b, hA, ROWS, DIMC, DIMC);

  // ---- stage 3: temporal attention (gathered via indexing; LN/GEMM are row-local) ----
  LN(hB, lnt_w, lnt_b, norm16);
  GEMMh(norm16, wqtt, q16, ROWS, DIMC, DIMC);
  GEMMh(norm16, wktt, k16, ROWS, DIMC, DIMC);
  GEMMh(norm16, wvtt, v16, ROWS, DIMC, DIMC);
  REP(q16, Qp, 3, 0, DSP, 16, 16);
  REP(k16, Kp, 3, 0, DSP, 64, 16);
  REP(v16, Vt, 3, 1, DSP, 64, 16);
  k_attn<<<dim3(DSP * NHEAD, 1), 32, 0, stream>>>(Qp, Kp, Vt, aO16,
                                                  16, 64, 16, ATTN_SCALE, 1);
  GEMMf(aO16, wott, hA, ot_b, hB, ROWS, DIMC, DIMC);

  // ---- stage 4: GEGLU feed-forward ----
  LN(hA, ln3_w, ln3_b, norm16);
  k_geglu<<<dim3(ROWS / 64, FFI / 64), 128, 0, stream>>>(norm16, wggt, gg_b, ffin);
  GEMMf(ffin, wfft, out, ffo_b, hA, ROWS, DIMC, FFI);
}
